// FullyConvAE_22720376996196
// MI455X (gfx1250) — matrix operand order repaired and checked by reading, not compiled, not run on hardware
//
#include <hip/hip_runtime.h>
#include <math.h>

// Problem constants (from reference): B=8, IN_PN=4096, OUT_PN=2048, M=16, W=32, IC=32, OC=64
#define BATCH  8
#define IN_PN  4096
#define OUT_PN 2048
#define MNB    16
#define WK     32
#define IC     32
#define OC     64

typedef __attribute__((ext_vector_type(16))) __bf16 v16bf;
typedef __attribute__((ext_vector_type(8)))  __bf16 v8bf;
typedef __attribute__((ext_vector_type(2)))  __bf16 v2bf;
typedef __attribute__((ext_vector_type(8)))  float  v8f;
typedef __attribute__((ext_vector_type(4)))  float  f4;
typedef __attribute__((ext_vector_type(4)))  unsigned int u4;

// pack two f32 into one dword of two bf16 (RNE via native fptrunc ->
// backend selects v_cvt_bf16_f32 / v_cvt_pk_bf16_f32)
__device__ __forceinline__ unsigned int pack2bf(float a, float b) {
    union { v2bf v; unsigned int u; } c;
    c.v = (v2bf){(__bf16)a, (__bf16)b};
    return c.u;
}
__device__ __forceinline__ float bflo2f(unsigned int d) { return __uint_as_float(d << 16); }
__device__ __forceinline__ float bfhi2f(unsigned int d) { return __uint_as_float(d & 0xFFFF0000u); }

union ABfrag { v16bf v; v8bf h[2]; };

__device__ __forceinline__ v8bf zero_v8bf() {
    union { u4 u; v8bf v; } z; z.u = (u4){0u, 0u, 0u, 0u}; return z.v;
}

// ---------------------------------------------------------------------------
// Prep: build bf16 B-matrices in d_ws.
//  conv:  K ordering is (i,w):  kernB[i*2048 + o*32 + w] = bf16(weights[w*2048 + o*32 + i])
//  res :  wresB[o*32 + i]       = bf16(weight_res[o*32 + i])
// ---------------------------------------------------------------------------
__global__ __launch_bounds__(256)
void prep_bf16(const float* __restrict__ weights, const float* __restrict__ wres,
               __bf16* __restrict__ ws) {
    int j = blockIdx.x * 256 + threadIdx.x;
    if (j < WK * OC * IC) {
        int i = j >> 11, rem = j & 2047;        // rem = o*32 + w
        int o = rem >> 5, w = rem & 31;
        ws[j] = (__bf16)weights[w * (OC * IC) + o * IC + i];
    } else if (j < WK * OC * IC + OC * IC) {
        ws[j] = (__bf16)wres[j - WK * OC * IC];
    }
}

// ---------------------------------------------------------------------------
// Fused kernel: one block (128 thr = 4 waves) = 16 output rows = 2 pts x 8 batches
//   row j -> point p0 + (j>>3), batch j&7 ; col = b*32+i
// Phase C (WMMA): X_pt[w, (b,i)] = sum_m Awm[pt][w][m] * Gt[pt][(b,i)][m], K=16 padded to 32
// Phase E (WMMA): OUT(16x64) = X(16x1024, k=(i,w)) * kernB(1024x64)  [32 k-chunks]
//                 RES(16x64) = g(16x32) * wresB(32x64)
// ---------------------------------------------------------------------------
__global__ __launch_bounds__(128)
void fullyconv_fused(const float* __restrict__ in_pc,
                     const float* __restrict__ bias,
                     const float* __restrict__ w_weights,
                     const float* __restrict__ p_raw,
                     const int*   __restrict__ neighbor_id,
                     const __bf16* __restrict__ kernB,   // d_ws
                     float* __restrict__ out) {
    __shared__ __align__(32) __bf16 Gt[2 * 256 * 32];   // [pt][col][m], m16..31 = 0 (32KB)
    __shared__ __align__(32) __bf16 Xl[16 * 1024];      // [row][i*32+w] (32KB)
    __shared__ __align__(32) __bf16 Awm[2 * WK * MNB];  // [pt][w][m] (2KB)
    __shared__ __align__(32) __bf16 gl[16 * IC];        // residual A (1KB)
    __shared__ float pns[2 * MNB];
    __shared__ int   nids[2 * MNB];

    const int tid = threadIdx.x;
    const int p0  = blockIdx.x * 2;

    // ---- A1: neighbor ids ----
    if (tid < 32) nids[tid] = neighbor_id[(p0 + (tid >> 4)) * MNB + (tid & 15)];
    __syncthreads();

    // ---- A1b: prefetch the 256 gathered feature rows (global_prefetch_b8) ----
    for (int k = 0; k < 2; ++k) {
        int j = tid + k * 128;                 // 0..255 = pm*8 + b
        int pm = j >> 3, b = j & 7;
        int id = nids[pm];
        if (id < IN_PN)
            __builtin_prefetch(in_pc + (size_t)(b * IN_PN + id) * IC, 0, 3);
    }

    // ---- A2: zero Gt's padded K half; build Awm (masked, transposed, bf16); raw pns ----
    for (int k = 0; k < 8; ++k) {
        int j = tid + k * 128;                 // 0..1023
        int c = j >> 1, half = j & 1;          // c = pt*256+col
        *(f4*)(Gt + c * 32 + 16 + half * 8) = (f4){0.f, 0.f, 0.f, 0.f};
    }
    for (int k = 0; k < 8; ++k) {
        int j = tid + k * 128;                 // 0..1023 = [pt][w][m]
        int pt = j >> 9, rem = j & 511;
        int w = rem >> 4, m = rem & 15;
        float msk = (nids[pt * MNB + m] != IN_PN) ? 1.f : 0.f;
        Awm[(pt * WK + w) * MNB + m] =
            (__bf16)(w_weights[((p0 + pt) * MNB + m) * WK + w] * msk);
    }
    if (tid < 32) {
        float msk = (nids[tid] != IN_PN) ? 1.f : 0.f;
        pns[tid] = fabsf(p_raw[(p0 + (tid >> 4)) * MNB + (tid & 15)]) * msk;
    }
    __syncthreads();

    // ---- A3: normalize p (2 lanes) ; B: gather neighbors -> Gt bf16 transposed ----
    if (tid < 2) {
        float s = 0.f;
        for (int m = 0; m < MNB; ++m) s += pns[tid * MNB + m];
        float inv = 1.f / (s + 1e-8f);
        for (int m = 0; m < MNB; ++m) pns[tid * MNB + m] *= inv;
    }
    for (int k = 0; k < 16; ++k) {
        int idx = tid + k * 128;               // 0..2047
        int m = idx & 15;                      // m in lane -> contiguous LDS halfword stores
        int ri = idx >> 4;                     // 0..127 = row*8 + l4
        int row = ri >> 3, l4 = ri & 7;
        int pt = row >> 3, b = row & 7;
        int id = nids[pt * MNB + m];
        f4 v = {0.f, 0.f, 0.f, 0.f};
        if (id < IN_PN)
            v = *(const f4*)(in_pc + (size_t)(b * IN_PN + id) * IC + l4 * 4);
        __bf16* dst = Gt + (pt * 256 + b * IC + l4 * 4) * 32 + m;
        dst[0]  = (__bf16)v.x;
        dst[32] = (__bf16)v.y;
        dst[64] = (__bf16)v.z;
        dst[96] = (__bf16)v.w;
    }
    __syncthreads();

    const int wv   = tid >> 5;
    const int lane = tid & 31;
    const int colT = lane & 15;                // tile-local row (A) / col (C,D)
    const int hi16 = (lane >= 16);
    const int ksA  = hi16 ? 8 : 0;             // 16-bit A layout: K halves per lane group
    const int selB = hi16 ? 16 : 0;            // unused: B is loaded with ksA, same K order as A

    // ---- C (WMMA): X_pt = Awm x Gt, write Xl[row][i*32+w] as packed b128 ----
    const v8bf zh = zero_v8bf();
    for (int pt = 0; pt < 2; ++pt) {
        for (int wt = 0; wt < 2; ++wt) {
            ABfrag a;
            a.h[0] = *(const v8bf*)(Awm + (pt * WK + wt * 16 + colT) * MNB + ksA);
            a.h[1] = zh;                       // K = m 16..23 / 24..31: structurally zero
            for (int nt = 0; nt < 4; ++nt) {
                int cb  = wv * 64 + nt * 16;
                int col = cb + colT;
                ABfrag bq;
                bq.h[0] = *(const v8bf*)(Gt + (pt * 256 + col) * 32 + ksA);
                bq.h[1] = *(const v8bf*)(Gt + (pt * 256 + col) * 32 + ksA + 16);
                v16bf bf = bq.v;
                v8f z = {0.f, 0.f, 0.f, 0.f, 0.f, 0.f, 0.f, 0.f};
                v8f c = __builtin_amdgcn_wmma_f32_16x16x32_bf16(
                    false, a.v, false, bf, (short)0, z, false, false);
                // c[v] -> w = wt*16 + (hi16?8:0) + v  (8 consecutive w), fixed (b,i)=col
                int b = col >> 5, i = col & 31;
                __bf16* dst =
                    Xl + (pt * 8 + b) * 1024 + i * 32 + wt * 16 + (hi16 ? 8 : 0);
                u4 pk;
                pk.x = pack2bf(c[0], c[1]);
                pk.y = pack2bf(c[2], c[3]);
                pk.z = pack2bf(c[4], c[5]);
                pk.w = pack2bf(c[6], c[7]);
                *(u4*)dst = pk;
            }
        }
    }

    // ---- D: residual A-matrix g[row][i] = sum_m pns[pt][m] * Gt[pt][col][m] ----
    for (int q = 0; q < 4; ++q) {
        int oidx = q * 128 + tid;              // 0..511
        int row = oidx >> 5, i = oidx & 31;
        int pt = row >> 3, b = row & 7;
        const unsigned int* gr = (const unsigned int*)(Gt + (pt * 256 + b * IC + i) * 32);
        float acc = 0.f;
#pragma unroll
        for (int m2 = 0; m2 < 8; ++m2) {
            unsigned int d = gr[m2];
            acc += pns[pt * MNB + 2 * m2]     * bflo2f(d);
            acc += pns[pt * MNB + 2 * m2 + 1] * bfhi2f(d);
        }
        gl[row * IC + i] = (__bf16)acc;
    }
    __syncthreads();

    // ---- E (WMMA): conv K=1024 (32 chunks, chunk = i) + residual K=32 ----
    const int o = wv * 16 + colT;

    v8f c_conv = {0.f, 0.f, 0.f, 0.f, 0.f, 0.f, 0.f, 0.f};
    v8f c_res  = {0.f, 0.f, 0.f, 0.f, 0.f, 0.f, 0.f, 0.f};
    {
        ABfrag a;
        a.h[0] = *(const v8bf*)(gl + colT * IC + ksA);
        a.h[1] = *(const v8bf*)(gl + colT * IC + ksA + 16);
        ABfrag bq;
        bq.h[0] = *(const v8bf*)(kernB + WK * OC * IC + o * IC + ksA);
        bq.h[1] = *(const v8bf*)(kernB + WK * OC * IC + o * IC + ksA + 16);
        v16bf b = bq.v;
        c_res = __builtin_amdgcn_wmma_f32_16x16x32_bf16(
            false, a.v, false, b, (short)0, c_res, false, false);
    }
    for (int ch = 0; ch < 32; ++ch) {          // chunk = i
        ABfrag a;
        a.h[0] = *(const v8bf*)(Xl + colT * 1024 + ch * 32 + ksA);
        a.h[1] = *(const v8bf*)(Xl + colT * 1024 + ch * 32 + ksA + 16);
        ABfrag bq;
        bq.h[0] = *(const v8bf*)(kernB + ch * (OC * IC) + o * IC + ksA);
        bq.h[1] = *(const v8bf*)(kernB + ch * (OC * IC) + o * IC + ksA + 16);
        v16bf b = bq.v;
        c_conv = __builtin_amdgcn_wmma_f32_16x16x32_bf16(
            false, a.v, false, b, (short)0, c_conv, false, false);
    }

    // ---- Epilogue: elu(conv+bias)*sqrt(1-r) + res*sqrt(r);  VGPR v -> row v / v+8
    const float S = 0.70710678118654752f;      // sqrt(0.5)
    const float bo = bias[o];
#pragma unroll
    for (int v = 0; v < 8; ++v) {
        int row = hi16 ? (v + 8) : v;
        int p = p0 + (row >> 3);
        int b = row & 7;
        float x = c_conv[v] + bo;
        float e = (x > 0.f) ? x : expm1f(x);
        out[((size_t)(b * OUT_PN + p)) * OC + o] = e * S + c_res[v] * S;
    }
}

// ---------------------------------------------------------------------------
extern "C" void kernel_launch(void* const* d_in, const int* in_sizes, int n_in,
                              void* d_out, int out_size, void* d_ws, size_t ws_size,
                              hipStream_t stream) {
    const float* in_pc   = (const float*)d_in[0];  // (B, IN_PN, IC)
    const float* weights = (const float*)d_in[1];  // (W, OC*IC)
    const float* bias    = (const float*)d_in[2];  // (OC,)
    const float* w_w     = (const float*)d_in[3];  // (OUT_PN, M, W)
    const float* p_raw   = (const float*)d_in[4];  // (OUT_PN, M)
    const float* wres    = (const float*)d_in[5];  // (OC, IC)
    const int*   nid     = (const int*)d_in[6];    // (OUT_PN, M)
    __bf16* ws           = (__bf16*)d_ws;          // bf16: 65536 kern + 2048 wres
    float* out           = (float*)d_out;          // (B, OUT_PN, OC)

    const int prep_elems = WK * OC * IC + OC * IC; // 67584
    prep_bf16<<<(prep_elems + 255) / 256, 256, 0, stream>>>(weights, wres, ws);
    fullyconv_fused<<<OUT_PN / 2, 128, 0, stream>>>(in_pc, bias, w_w, p_raw, nid, ws, out);
}
